// SubLSTM_71167608095137
// MI455X (gfx1250) — matrix operand order repaired and checked by reading, not compiled, not run on hardware
//
#include <hip/hip_runtime.h>
#include <stdint.h>

// ---------------------------------------------------------------------------
// SubLSTM (2 layers, T=512, B=32, I=H=1024) for gfx1250 (MI455X, wave32, WMMA)
//
// Structure: persistent scan kernel, 16 blocks x 256 threads. The input
// projection is fused into the per-step GEMM (K=2048 total), bf16 WMMA with
// f32 accumulation, weights pre-converted to bf16 (L2-resident, 32 MB),
// c-state lives in VGPRs for the entire 512-step scan, one device-wide
// barrier per step. h(t-1) is staged into LDS per step via
// GLOBAL_LOAD_ASYNC_TO_LDS_B128 (ASYNCcnt / s_wait_asynccnt).
// ---------------------------------------------------------------------------

#define T_STEPS 512
#define BATCH   32
#define HDIM    1024
#define GDIM    4096      // 4*H
#define NBLK    16        // persistent blocks in the scan kernel

// Flip to disable the async-DMA staging path (falls back to b128 copy loop).
#define USE_ASYNC_LDS 1

typedef __attribute__((ext_vector_type(16))) __bf16 v16bf;
typedef __attribute__((ext_vector_type(8)))  float  v8f;

// round-to-nearest-even f32 -> bf16 (stored as ushort)
__device__ __forceinline__ unsigned short f2bf(float f) {
  union { float f; unsigned u; } x; x.f = f;
  unsigned u = x.u;
  unsigned r = (u + 0x7fffu + ((u >> 16) & 1u)) >> 16;
  return (unsigned short)r;
}

__device__ __forceinline__ float sigmoidf(float x) {
  return 1.0f / (1.0f + __expf(-x));
}

// ---------------------------------------------------------------------------
// f32 -> bf16 bulk conversion
// ---------------------------------------------------------------------------
__global__ void cvt_f32_bf16(const float* __restrict__ in,
                             unsigned short* __restrict__ out, int n) {
  int i = blockIdx.x * blockDim.x + threadIdx.x;
  int stride = gridDim.x * blockDim.x;
  for (; i < n; i += stride) out[i] = f2bf(in[i]);
}

// ---------------------------------------------------------------------------
// Grid-wide barrier: HW cluster barrier if dispatched as a cluster, else
// atomic arrival counter + generation spin (s_sleep while waiting).
// ---------------------------------------------------------------------------
__device__ __forceinline__ void grid_barrier(unsigned* bar, unsigned nblk, int iter) {
  __threadfence();
  __syncthreads();
  if (__builtin_amdgcn_cluster_id_x() != 0) {
    // launched as one 16-WG cluster: s_barrier_signal/-wait -3
    if (threadIdx.x < 32) __builtin_amdgcn_s_cluster_barrier();
  } else if (threadIdx.x == 0) {
    atomicAdd(bar, 1u);
    unsigned target = nblk * (unsigned)(iter + 1);
    while (*(volatile unsigned*)bar < target) {
      __builtin_amdgcn_s_sleep(2);
    }
  }
  __syncthreads();
  __threadfence();
}

// ---------------------------------------------------------------------------
// Persistent scan kernel (one subLSTM layer).
//   a_in : [T,32,1024]  bf16 input sequence (x, or previous layer's h chain)
//   w_ih : [4096,1024]  bf16   (gates = a_in@w_ihT + h@w_hhT + bias)
//   w_hh : [4096,1024]  bf16
//   bias : [4096]       f32
//   hbuf : [T+1,32,1024] bf16, row 0 pre-zeroed; row t+1 = h(t)
//   out  : [T,32,1024]  f32 or nullptr
//
// 16 blocks x 256 threads (8 waves). Block jb owns 64 H-columns.
// wave w: mt = w&1 (batch rows 16*mt..), jg = w>>1 (16-column j group).
// Each wave keeps 4 gate accumulators (i,o,z,f) + its c slice in VGPRs.
// ---------------------------------------------------------------------------
__launch_bounds__(256, 1)
__global__ void sublstm_scan(const unsigned short* __restrict__ a_in,
                             const unsigned short* __restrict__ w_ih,
                             const unsigned short* __restrict__ w_hh,
                             const float* __restrict__ bias,
                             unsigned short* __restrict__ hbuf,
                             float* __restrict__ out_f32,
                             unsigned* __restrict__ bar) {
  __shared__ unsigned short h_lds[BATCH * HDIM];      // 64 KB: h(t-1) in bf16

  const int tid   = threadIdx.x;
  const int lane  = tid & 31;
  const int wave  = tid >> 5;
  const int mt    = wave & 1;                         // batch tile (0/1)
  const int jg    = wave >> 1;                        // 0..3
  const int jb    = blockIdx.x;                       // 0..15
  const int lhalf = lane >> 4;                        // 0 / 1
  const int jcol  = jb * 64 + jg * 16 + (lane & 15);  // this lane's H column

  // A-fragment K offsets (16-bit A 16x32 layout): lanes 0-15 -> K {0..7,16..23},
  // lanes 16-31 -> K {8..15,24..31}
  const int a_off = lhalf ? 8 : 0;
  // B-fragment: same K split as A (lanes 0-15 -> K {0..7,16..23}, 16-31 -> {8..15,24..31})
  const int b_off = lhalf ? 8 : 0;
  const int arow  = mt * 16 + (lane & 15);            // batch row of A fragment

  // Per-gate weight row pointers (column jcol of B == row g of W, contiguous K)
  const unsigned short* wih_g[4];
  const unsigned short* whh_g[4];
  float bias_g[4];
#pragma unroll
  for (int g = 0; g < 4; ++g) {
    const int grow = g * HDIM + jcol;
    wih_g[g] = w_ih + (size_t)grow * HDIM + b_off;
    whh_g[g] = w_hh + (size_t)grow * HDIM + b_off;
    bias_g[g] = bias[grow];
  }

  v8f c_state = {0.f, 0.f, 0.f, 0.f, 0.f, 0.f, 0.f, 0.f};

  for (int t = 0; t < T_STEPS; ++t) {
    // ---- stage h(t-1) (= hbuf row t) into LDS ----
    {
      const unsigned short* src = hbuf + (size_t)t * BATCH * HDIM;
#if USE_ASYNC_LDS
      // gfx1250 async DMA straight into LDS: 256 threads x 16 x b128 = 64 KB.
      // LDS byte address = low 32 bits of the generic shared pointer
      // (flat-LDS aperture truncates to the wave's LDS offset).
      const unsigned lds_base = (unsigned)(uintptr_t)h_lds + (unsigned)tid * 16u;
      const char* gbase = (const char*)src + (size_t)tid * 16u;
#pragma unroll
      for (int i = 0; i < 16; ++i) {
        asm volatile("global_load_async_to_lds_b128 %0, %1, off"
                     :: "v"(lds_base + (unsigned)i * 4096u),
                        "v"(gbase + (size_t)i * 4096u)
                     : "memory");
      }
      asm volatile("s_wait_asynccnt 0x0" ::: "memory");
#else
      const uint4* s4 = (const uint4*)src;
      uint4* d4 = (uint4*)h_lds;
#pragma unroll
      for (int i = 0; i < (BATCH * HDIM * 2) / (16 * 256); ++i)
        d4[tid + i * 256] = s4[tid + i * 256];
#endif
    }
    __syncthreads();

    v8f acc[4];
#pragma unroll
    for (int g = 0; g < 4; ++g) acc[g] = (v8f){0.f,0.f,0.f,0.f,0.f,0.f,0.f,0.f};

    // ---- K loop 1: input projection, A = a_in[t] (bf16, global/L2) ----
    {
      const unsigned short* aptr = a_in + ((size_t)t * BATCH + arow) * HDIM;
      for (int kc = 0; kc < HDIM / 32; ++kc) {
        const int k0 = kc * 32;
        union { uint4 u[2]; v16bf v; } af;
        af.u[0] = *(const uint4*)(aptr + k0 + a_off);
        af.u[1] = *(const uint4*)(aptr + k0 + a_off + 16);
#pragma unroll
        for (int g = 0; g < 4; ++g) {
          union { uint4 u[2]; v16bf v; } bfm;
          const unsigned short* bp = wih_g[g] + k0;
          bfm.u[0] = *(const uint4*)(bp);
          bfm.u[1] = *(const uint4*)(bp + 16);
          acc[g] = __builtin_amdgcn_wmma_f32_16x16x32_bf16(
              false, af.v, false, bfm.v, (short)0, acc[g], false, false);
        }
      }
    }

    // ---- K loop 2: recurrent projection, A = h(t-1) from LDS ----
    {
      const unsigned short* hptr = h_lds + arow * HDIM;
      for (int kc = 0; kc < HDIM / 32; ++kc) {
        const int k0 = kc * 32;
        union { uint4 u[2]; v16bf v; } af;
        af.u[0] = *(const uint4*)(hptr + k0 + a_off);
        af.u[1] = *(const uint4*)(hptr + k0 + a_off + 16);
#pragma unroll
        for (int g = 0; g < 4; ++g) {
          union { uint4 u[2]; v16bf v; } bfm;
          const unsigned short* bp = whh_g[g] + k0;
          bfm.u[0] = *(const uint4*)(bp);
          bfm.u[1] = *(const uint4*)(bp + 16);
          acc[g] = __builtin_amdgcn_wmma_f32_16x16x32_bf16(
              false, af.v, false, bfm.v, (short)0, acc[g], false, false);
        }
      }
    }

    // ---- speculative prefetch of next step's A row (global_prefetch_b8) ----
    if (t + 1 < T_STEPS) {
      const char* nxt =
          (const char*)(a_in + ((size_t)(t + 1) * BATCH + arow) * HDIM);
      __builtin_prefetch(nxt + lhalf * 1024, 0, 1);
      __builtin_prefetch(nxt + lhalf * 1024 + 512, 0, 1);
    }

    // ---- gates -> c,h update (C/D layout: row = r + 8*lhalf, col = lane&15) --
#pragma unroll
    for (int r = 0; r < 8; ++r) {
      const float gi = sigmoidf(acc[0][r] + bias_g[0]);
      const float go = sigmoidf(acc[1][r] + bias_g[1]);
      const float gz = sigmoidf(acc[2][r] + bias_g[2]);
      const float gf = sigmoidf(acc[3][r] + bias_g[3]);
      const float cv = c_state[r] * gf + gz - gi;
      c_state[r] = cv;
      const float hv = sigmoidf(cv) - go;

      const int brow = mt * 16 + lhalf * 8 + r;
      hbuf[((size_t)(t + 1) * BATCH + brow) * HDIM + jcol] = f2bf(hv);
      if (out_f32)
        out_f32[((size_t)t * BATCH + brow) * HDIM + jcol] = hv;
    }

    // ---- device-wide step barrier ----
    grid_barrier(bar, NBLK, t);
  }
}

// ---------------------------------------------------------------------------
// Host launcher
// ---------------------------------------------------------------------------
extern "C" void kernel_launch(void* const* d_in, const int* in_sizes, int n_in,
                              void* d_out, int out_size, void* d_ws, size_t ws_size,
                              hipStream_t stream) {
  (void)in_sizes; (void)n_in; (void)out_size; (void)ws_size;

  const float* x    = (const float*)d_in[0];
  const float* wih0 = (const float*)d_in[1];
  const float* whh0 = (const float*)d_in[2];
  const float* b0   = (const float*)d_in[3];
  const float* wih1 = (const float*)d_in[4];
  const float* whh1 = (const float*)d_in[5];
  const float* b1   = (const float*)d_in[6];
  float* out = (float*)d_out;

  char* ws = (char*)d_ws;
  size_t off = 0;
  auto alloc = [&](size_t bytes) -> void* {
    void* p = ws + off;
    off += (bytes + 255) & ~(size_t)255;
    return p;
  };

  unsigned short* xbf    = (unsigned short*)alloc((size_t)T_STEPS * BATCH * HDIM * 2);
  unsigned short* wih0bf = (unsigned short*)alloc((size_t)GDIM * HDIM * 2);
  unsigned short* whh0bf = (unsigned short*)alloc((size_t)GDIM * HDIM * 2);
  unsigned short* wih1bf = (unsigned short*)alloc((size_t)GDIM * HDIM * 2);
  unsigned short* whh1bf = (unsigned short*)alloc((size_t)GDIM * HDIM * 2);
  unsigned short* h0     = (unsigned short*)alloc((size_t)(T_STEPS + 1) * BATCH * HDIM * 2);
  unsigned short* h1     = (unsigned short*)alloc((size_t)(T_STEPS + 1) * BATCH * HDIM * 2);
  unsigned*       bar    = (unsigned*)alloc(256);

  // bf16 copies (weights become L2-resident for the whole scan)
  cvt_f32_bf16<<<4096, 256, 0, stream>>>(x,    xbf,    T_STEPS * BATCH * HDIM);
  cvt_f32_bf16<<<1024, 256, 0, stream>>>(wih0, wih0bf, GDIM * HDIM);
  cvt_f32_bf16<<<1024, 256, 0, stream>>>(whh0, whh0bf, GDIM * HDIM);
  cvt_f32_bf16<<<1024, 256, 0, stream>>>(wih1, wih1bf, GDIM * HDIM);
  cvt_f32_bf16<<<1024, 256, 0, stream>>>(whh1, whh1bf, GDIM * HDIM);

  // zero h(-1) rows and barrier counters (re-done every call: graph-safe,
  // deterministic across replays)
  hipMemsetAsync(h0, 0, BATCH * HDIM * 2, stream);
  hipMemsetAsync(h1, 0, BATCH * HDIM * 2, stream);
  hipMemsetAsync(bar, 0, 256, stream);

  // layer 0: input = x, no f32 output needed (layer 1 reads the bf16 h chain)
  sublstm_scan<<<NBLK, 256, 0, stream>>>(xbf, wih0bf, whh0bf, b0, h0, nullptr, bar);
  // layer 1: input = layer-0 h chain rows 1..T; writes final f32 output
  sublstm_scan<<<NBLK, 256, 0, stream>>>(h0 + (size_t)BATCH * HDIM, wih1bf, whh1bf,
                                         b1, h1, out, bar + 32);
}
